// DXVAE_2370821947804
// MI455X (gfx1250) — compile-verified, matrix operand order checked by reading, not run on hardware
//
#include <hip/hip_runtime.h>
#include <stdint.h>
#include <stddef.h>

// ---------------- problem dims ----------------
#define B_   2048
#define NN_  7
#define P_   12
#define PP_  32     // P padded to K%32==0 for WMMA
#define H_   512
#define H2_  1024
#define H3_  1536
#define H4_  2048
#define L_   128
#define EPS_ 1e-7f

typedef unsigned short bf16u;
typedef __attribute__((ext_vector_type(16))) __bf16 v16bf;
typedef __attribute__((ext_vector_type(8)))  float  v8f;

enum { ACT_NONE=0, ACT_SIG=1, ACT_TANH=2, ACT_RELU=3, ACT_SP=4 };

#define CEILDIV(a,b) (((a)+(b)-1)/(b))

// ---------------- device helpers ----------------
__device__ __forceinline__ bf16u f2bf(float f){
  union { float f; uint32_t u; } v; v.f = f;
  uint32_t u = v.u;
  u += 0x7fffu + ((u >> 16) & 1u);          // round-to-nearest-even
  return (bf16u)(u >> 16);
}
__device__ __forceinline__ float sigmf(float x){ return 1.0f/(1.0f+__expf(-x)); }
__device__ __forceinline__ float bcef(float p, float t){
  p = fminf(fmaxf(p, EPS_), 1.0f-EPS_);
  return -(t*__logf(p) + (1.0f-t)*__logf(1.0f-p));
}
__device__ __forceinline__ float blockReduce256(float v){
  __shared__ float s[256];
  s[threadIdx.x] = v; __syncthreads();
  for (int st = 128; st > 0; st >>= 1){
    if ((int)threadIdx.x < st) s[threadIdx.x] += s[threadIdx.x + st];
    __syncthreads();
  }
  return s[0];
}

// ---------------- WMMA GEMM: C = act(A(MxK,bf16) * W(NxK,bf16)^T + bias) ----------------
// A row-major lda; W row-major ldw (weight layout (out,in) => computes x@W.T directly).
// For v_wmma_f32_16x16x32_bf16 the B-operand (=W^T) fragment layout equals the A-operand
// layout applied to row-major W, so both fragments load identically: per lane,
// row = lane&15 (M for A / N for W), K-halves selected by lane>=16 (ko = 0 or 8),
// u[0..3] = K[ko..ko+7], u[4..7] = K[16+ko..16+ko+7]  (8 dwords/lane).
//
// Wave tile 32x64 (2 M-frags x 4 N-frags): 8 WMMAs per 6 fragment loads per K-step of 32.
// Block = 8 waves (4 along M, 2 along N) => block tile 128x128. Data is L2-resident
// (192MB L2 holds all weights+activations) so direct global loads feed the WMMA pipe.
union FragU { v16bf v; uint32_t u[8]; };

__device__ __forceinline__ v16bf load_frag(const uint32_t* __restrict__ row, int dw){
  FragU f;
#pragma unroll
  for (int i=0;i<4;i++) f.u[i]   = row[dw+i];
#pragma unroll
  for (int i=0;i<4;i++) f.u[4+i] = row[dw+8+i];
  return f.v;
}

template<int ACT>
__global__ __launch_bounds__(256) void k_gemm(
    const bf16u* __restrict__ A, int lda,
    const bf16u* __restrict__ W, int ldw,
    const float* __restrict__ bias,
    float* __restrict__ C, int ldc,
    bf16u* __restrict__ Cbf,
    int M, int Nreal, int K)
{
  const int lane  = threadIdx.x & 31;
  const int wave  = threadIdx.x >> 5;
  const int m0    = blockIdx.y*128 + (wave & 3)*32;   // M multiple of 128 guaranteed by host
  const int n0    = blockIdx.x*128 + (wave >> 2)*64;  // Nw multiple of 128 (weights zero-padded)
  const int row   = lane & 15;
  const int koSel = (lane >> 4) << 3;                 // 0 or 8 bf16 elements

  const uint32_t* ap[2];
  const uint32_t* bp[4];
#pragma unroll
  for (int i=0;i<2;i++) ap[i] = (const uint32_t*)(A + (size_t)(m0 + 16*i + row)*lda);
#pragma unroll
  for (int j=0;j<4;j++) bp[j] = (const uint32_t*)(W + (size_t)(n0 + 16*j + row)*ldw);

  v8f acc[2][4] = {};
  for (int k0 = 0; k0 < K; k0 += 32) {
    const int dw = (k0 + koSel) >> 1;
    v16bf fa[2], fb[4];
#pragma unroll
    for (int i=0;i<2;i++) fa[i] = load_frag(ap[i], dw);
#pragma unroll
    for (int j=0;j<4;j++) fb[j] = load_frag(bp[j], dw);
#pragma unroll
    for (int i=0;i<2;i++)
#pragma unroll
      for (int j=0;j<4;j++)
        acc[i][j] = __builtin_amdgcn_wmma_f32_16x16x32_bf16(
            false, fa[i], false, fb[j], (short)0, acc[i][j], false, false);
  }

  auto epi = [&](const v8f& a, int mb, int nb){
    const int n = nb + (lane & 15);
    if (n >= Nreal) return;
    const int mtop = mb + ((lane >> 4) << 3);
    const float bv = bias ? bias[n] : 0.0f;
#pragma unroll
    for (int r = 0; r < 8; r++){
      float x = a[r] + bv;
      if (ACT == ACT_SIG)  x = sigmf(x);
      if (ACT == ACT_TANH) x = tanhf(x);
      if (ACT == ACT_RELU) x = x > 0.0f ? x : 0.0f;
      if (ACT == ACT_SP)   x = (x > 20.0f) ? x : log1pf(__expf(x));
      size_t o = (size_t)(mtop + r)*ldc + n;
      C[o] = x;
      if (Cbf) Cbf[o] = f2bf(x);
    }
  };
#pragma unroll
  for (int i=0;i<2;i++)
#pragma unroll
    for (int j=0;j<4;j++)
      epi(acc[i][j], m0 + 16*i, n0 + 16*j);
}

// ---------------- elementwise / builder kernels ----------------
__global__ void k_zero32(uint32_t* p, size_t n){
  size_t i = (size_t)blockIdx.x*256 + threadIdx.x;
  if (i < n) p[i] = 0u;
}

// f32 -> bf16 with row/col zero padding (generic weight staging)
__global__ void k_cvt_pad(const float* __restrict__ in, bf16u* __restrict__ out,
                          int rin, int cin, int rout, int cout){
  size_t i = (size_t)blockIdx.x*256 + threadIdx.x;
  size_t n = (size_t)rout*cout;
  if (i >= n) return;
  int r = (int)(i / cout), c = (int)(i % cout);
  float v = (r < rin && c < cin) ? in[(size_t)r*cin + c] : 0.0f;
  out[i] = f2bf(v);
}

// Build X (B x PP_) bf16 from params[:,v,:]; maskmode 0 -> 1.0, 1 -> adj[b][v][v]
__global__ void k_build_x(const float* __restrict__ params, const int* __restrict__ adj,
                          bf16u* __restrict__ out, int v, int maskmode){
  int i = blockIdx.x*256 + threadIdx.x;
  if (i >= B_*PP_) return;
  int b = i / PP_, c = i % PP_;
  float val = 0.0f;
  if (c < P_) {
    float m = maskmode ? (float)adj[(size_t)b*(NN_*NN_) + v*NN_ + v] : 1.0f;
    val = params[((size_t)b*NN_ + v)*P_ + c] * m;
  }
  out[i] = f2bf(val);
}

// Hcat builder (unified encoder/decoder):
//   node = nodeBase + kk; masked by (kk >= thr)
//   m_in = adj[b][node][vc], m_out = adj[b][vc][node]
//   hcat[(b*k+kk)][0:H)   = hid[node][b][:] * m_in
//   hcat[(b*k+kk)][H:2H)  = hid[node][b][:] * m_out
__global__ void k_build_hcat(const float* __restrict__ hid, const int* __restrict__ adj,
                             bf16u* __restrict__ hcat, int vc, int k, int nodeBase, int thr){
  int i = blockIdx.x*256 + threadIdx.x;
  int tot = B_*k*H_;
  if (i >= tot) return;
  int b  = i / (k*H_);
  int rm = i % (k*H_);
  int kk = rm / H_;
  int j  = rm % H_;
  int node = nodeBase + kk;
  float hv = hid[((size_t)node*B_ + b)*H_ + j];
  float mi = 0.0f, mo = 0.0f;
  if (kk >= thr) {
    mi = (float)adj[(size_t)b*(NN_*NN_) + node*NN_ + vc];
    mo = (float)adj[(size_t)b*(NN_*NN_) + vc*NN_ + node];
  }
  size_t rowO = ((size_t)b*k + kk)*H2_;
  hcat[rowO + j]      = f2bf(hv*mi);
  hcat[rowO + H_ + j] = f2bf(hv*mo);
}

// Hin[b][j] = sum_kk sigmoid(gate)[..]*map[..]  (gate already sigmoided in GEMM epilogue)
__global__ void k_gm_reduce(const float* __restrict__ gp, const float* __restrict__ mp,
                            int k, float* __restrict__ hout, bf16u* __restrict__ houtb){
  int i = blockIdx.x*256 + threadIdx.x;
  if (i >= B_*H_) return;
  int b = i / H_, j = i % H_;
  float s = 0.0f;
  for (int kk = 0; kk < k; kk++){
    size_t r = ((size_t)b*k + kk)*H_ + j;
    s += gp[r]*mp[r];
  }
  hout[i] = s; houtb[i] = f2bf(s);
}

// GRU combine: h' = (1-z)*n + z*h
__global__ void k_gru_combine(const float* __restrict__ gi, const float* __restrict__ gh,
                              const float* __restrict__ hold,
                              float* __restrict__ hnew, bf16u* __restrict__ hnewb){
  int i = blockIdx.x*256 + threadIdx.x;
  if (i >= B_*H_) return;
  int b = i / H_, j = i % H_;
  const float* gib = gi + (size_t)b*H3_;
  const float* ghb = gh + (size_t)b*H3_;
  float r = sigmf(gib[j]        + ghb[j]);
  float z = sigmf(gib[H_+j]     + ghb[H_+j]);
  float n = tanhf(gib[2*H_+j] + r*ghb[2*H_+j]);
  float h = hold[i];
  float o = (1.0f - z)*n + z*h;
  hnew[i] = o; hnewb[i] = f2bf(o);
}

__global__ void k_build_concat(const bf16u* __restrict__ a, const bf16u* __restrict__ b2,
                               bf16u* __restrict__ out){
  int i = blockIdx.x*256 + threadIdx.x;
  if (i >= B_*H2_) return;
  int b = i / H2_, j = i % H2_;
  out[i] = (j < H_) ? a[(size_t)b*H_ + j] : b2[(size_t)b*H_ + (j - H_)];
}

// ---------------- loss kernels (single block => deterministic) ----------------
__global__ void k_loss_lp(const float* __restrict__ Xi, const float* __restrict__ params,
                          int vi, float* __restrict__ acc){
  float s = 0.0f;
  for (int b = threadIdx.x; b < B_; b += 256){
    const float* xr = Xi + (size_t)b*16;
    const float* tr = params + ((size_t)b*NN_ + vi)*P_;
    float m = 0.0f;
#pragma unroll
    for (int c = 0; c < 11; c++){ float d = xr[c]-tr[c]; m += d*d; }
    s += m + 10.0f * bcef(sigmf(xr[11]), tr[11]);
  }
  float tot = blockReduce256(s);
  if (threadIdx.x == 0) acc[0] += tot / (float)B_;
}

__global__ void k_loss_es(const float* __restrict__ Es, const int* __restrict__ adj,
                          int vi, float* __restrict__ acc){
  float s = 0.0f;
  for (int b = threadIdx.x; b < B_; b += 256){
    float t = (float)adj[(size_t)b*(NN_*NN_) + vi*NN_ + vi];
    s += bcef(Es[(size_t)b*16], t);
  }
  float tot = blockReduce256(s);
  if (threadIdx.x == 0) acc[1] += tot / (float)B_;
}

__global__ void k_loss_edge(const float* __restrict__ e, const int* __restrict__ adj,
                            int vi, int vj, float* __restrict__ acc){
  float s = 0.0f;
  for (int b = threadIdx.x; b < B_; b += 256){
    float t0 = (float)adj[(size_t)b*(NN_*NN_) + vj*NN_ + vi];
    float t1 = (float)adj[(size_t)b*(NN_*NN_) + vi*NN_ + vj];
    s += bcef(e[(size_t)b*16], t0) + bcef(e[(size_t)b*16 + 1], t1);
  }
  float tot = blockReduce256(s);
  if (threadIdx.x == 0) acc[1] += tot / (float)B_;
}

__global__ void k_loss_kld(const float* __restrict__ mu, const float* __restrict__ sd,
                           float* __restrict__ acc){
  float s = 0.0f;
  for (int i = threadIdx.x; i < B_*L_; i += 256){
    float m = mu[i], d = sd[i];
    s += __logf(d) + (1.0f + m*m)/(2.0f*d*d) - 0.5f;
  }
  float tot = blockReduce256(s);
  if (threadIdx.x == 0) acc[2] += tot / (float)B_;
}

__global__ void k_finalize(const float* __restrict__ acc, float* __restrict__ out){
  if (threadIdx.x == 0){
    float lp = acc[0]*0.2f;   // W2_
    float le = acc[1]*1.0f;   // W3_
    float kw = acc[2]*1.0f;   // W4_
    out[0] = lp + le + kw; out[1] = lp; out[2] = le; out[3] = kw;
  }
}

// ---------------- host orchestration ----------------
static inline void gemm_launch(hipStream_t st, int act,
    const bf16u* A, int lda, const bf16u* W, int ldw, const float* bias,
    float* C, int ldc, bf16u* Cbf, int M, int Nw, int Nreal, int K)
{
  dim3 g(Nw/128, M/128), b(256,1,1);
  switch (act) {
    case ACT_NONE: k_gemm<ACT_NONE><<<g,b,0,st>>>(A,lda,W,ldw,bias,C,ldc,Cbf,M,Nreal,K); break;
    case ACT_SIG:  k_gemm<ACT_SIG ><<<g,b,0,st>>>(A,lda,W,ldw,bias,C,ldc,Cbf,M,Nreal,K); break;
    case ACT_TANH: k_gemm<ACT_TANH><<<g,b,0,st>>>(A,lda,W,ldw,bias,C,ldc,Cbf,M,Nreal,K); break;
    case ACT_RELU: k_gemm<ACT_RELU><<<g,b,0,st>>>(A,lda,W,ldw,bias,C,ldc,Cbf,M,Nreal,K); break;
    case ACT_SP:   k_gemm<ACT_SP  ><<<g,b,0,st>>>(A,lda,W,ldw,bias,C,ldc,Cbf,M,Nreal,K); break;
  }
}

extern "C" void kernel_launch(void* const* d_in, const int* in_sizes, int n_in,
                              void* d_out, int out_size, void* d_ws, size_t ws_size,
                              hipStream_t stream) {
  (void)in_sizes; (void)n_in; (void)out_size; (void)ws_size;

  // Inputs flattened in setup_inputs() dict insertion order:
  // 0: params (B,N,P) f32; 1: adj (B,N,N) i32;
  // 2..5: pe (Wih,Whh,bih,bhh); 6..9: le; 10..13: pd; 14..17: ld;
  // 18/19 mu; 20/21 std; 22/23 zh; 24/25 p1; 26/27 p2; 28/29 es1; 30/31 es2;
  // 32/33 e1; 34/35 e2; 36/37 gate; 38 map
  const float* params = (const float*)d_in[0];
  const int*   adj    = (const int*)d_in[1];

  // ---- workspace bump allocator (stateless per call) ----
  char* base = (char*)d_ws; size_t off = 0;
  auto takeF = [&](size_t n)->float* { off = (off+255) & ~(size_t)255; float* r=(float*)(base+off); off += n*4; return r; };
  auto takeB = [&](size_t n)->bf16u* { off = (off+255) & ~(size_t)255; bf16u* r=(bf16u*)(base+off); off += n*2; return r; };

  bf16u *wih[4], *whh[4];
  for (int g=0; g<4; g++){ wih[g]=takeB((size_t)H3_*PP_); whh[g]=takeB((size_t)H3_*H_); }
  bf16u* wMu  = takeB((size_t)L_*H_);   bf16u* wStd = takeB((size_t)L_*H_);
  bf16u* wZh  = takeB((size_t)H_*L_);
  bf16u* wP1  = takeB((size_t)H2_*H_);  bf16u* wP2  = takeB((size_t)128*H2_);
  bf16u* wEs1 = takeB((size_t)H2_*H_);  bf16u* wEs2 = takeB((size_t)128*H2_);
  bf16u* wE1  = takeB((size_t)H4_*H2_); bf16u* wE2  = takeB((size_t)128*H4_);
  bf16u* wGate= takeB((size_t)H_*H2_);  bf16u* wMap = takeB((size_t)H_*H2_);

  float* henc = takeF((size_t)NN_*B_*H_); bf16u* hencB = takeB((size_t)NN_*B_*H_);
  float* hdec = takeF((size_t)NN_*B_*H_); bf16u* hdecB = takeB((size_t)NN_*B_*H_);
  float* gi   = takeF((size_t)B_*H3_);    float* gh    = takeF((size_t)B_*H3_);
  bf16u* hcat = takeB((size_t)B_*6*H2_);
  float* gpre = takeF((size_t)B_*6*H_);   float* mpre  = takeF((size_t)B_*6*H_);
  float* gmH  = takeF((size_t)B_*H_);     bf16u* gmHB  = takeB((size_t)B_*H_);
  float* tmpH = takeF((size_t)B_*H_);     bf16u* tmpHB = takeB((size_t)B_*H_);
  float* hz   = takeF((size_t)B_*H_);     bf16u* hzB   = takeB((size_t)B_*H_);
  bf16u* xbuf = takeB((size_t)B_*PP_);    bf16u* xzero = takeB((size_t)B_*PP_);
  float* muF  = takeF((size_t)B_*L_);     bf16u* muB   = takeB((size_t)B_*L_);
  float* stdF = takeF((size_t)B_*L_);
  float* hinit= takeF((size_t)B_*H_);     bf16u* hinitB= takeB((size_t)B_*H_);
  float* t1   = takeF((size_t)B_*H2_);    bf16u* t1B   = takeB((size_t)B_*H2_);
  float* t2   = takeF((size_t)B_*H4_);    bf16u* t2B   = takeB((size_t)B_*H4_);
  bf16u* cc   = takeB((size_t)B_*H2_);
  float* xi   = takeF((size_t)B_*16);     float* esb   = takeF((size_t)B_*16);
  float* eb   = takeF((size_t)B_*16);
  float* acc  = takeF(4);

  // ---- zero init (loss accumulators, zero buffers) ----
  auto zero32 = [&](void* p, size_t dwords){
    k_zero32<<<(unsigned)CEILDIV(dwords,256),256,0,stream>>>((uint32_t*)p, dwords);
  };
  zero32(acc, 4);
  zero32(hz,  (size_t)B_*H_);
  zero32(hzB, (size_t)B_*H_/2);
  zero32(xzero,(size_t)B_*PP_/2);

  // ---- stage weights to bf16 (zero-padded where needed) ----
  auto cvt = [&](const void* in, bf16u* out, int rin, int cin, int rout, int cout){
    size_t n = (size_t)rout*cout;
    k_cvt_pad<<<(unsigned)CEILDIV(n,256),256,0,stream>>>((const float*)in, out, rin, cin, rout, cout);
  };
  const int gidx[4] = {2, 6, 10, 14};   // pe, le, pd, ld
  const float *bih[4], *bhh[4];
  for (int g=0; g<4; g++){
    cvt(d_in[gidx[g]+0], wih[g], H3_, P_, H3_, PP_);
    cvt(d_in[gidx[g]+1], whh[g], H3_, H_, H3_, H_);
    bih[g] = (const float*)d_in[gidx[g]+2];
    bhh[g] = (const float*)d_in[gidx[g]+3];
  }
  cvt(d_in[18], wMu,  L_,  H_,  L_,  H_ ); const float* bMu  = (const float*)d_in[19];
  cvt(d_in[20], wStd, L_,  H_,  L_,  H_ ); const float* bStd = (const float*)d_in[21];
  cvt(d_in[22], wZh,  H_,  L_,  H_,  L_ ); const float* bZh  = (const float*)d_in[23];
  cvt(d_in[24], wP1,  H2_, H_,  H2_, H_ ); const float* bP1  = (const float*)d_in[25];
  cvt(d_in[26], wP2,  P_,  H2_, 128, H2_); const float* bP2  = (const float*)d_in[27];
  cvt(d_in[28], wEs1, H2_, H_,  H2_, H_ ); const float* bEs1 = (const float*)d_in[29];
  cvt(d_in[30], wEs2, 1,   H2_, 128, H2_); const float* bEs2 = (const float*)d_in[31];
  cvt(d_in[32], wE1,  H4_, H2_, H4_, H2_); const float* bE1  = (const float*)d_in[33];
  cvt(d_in[34], wE2,  2,   H4_, 128, H4_); const float* bE2  = (const float*)d_in[35];
  cvt(d_in[36], wGate,H_,  H2_, H_,  H2_); const float* bGate= (const float*)d_in[37];
  cvt(d_in[38], wMap, H_,  H2_, H_,  H2_);

  const int EW = 256;
  auto buildX = [&](int v, int maskmode){
    k_build_x<<<CEILDIV(B_*PP_,EW),EW,0,stream>>>(params, adj, xbuf, v, maskmode);
  };
  // GRU: h' from (x, h_in); g: 0=pe 1=le 2=pd 3=ld
  auto gru = [&](const bf16u* xb, int g, const bf16u* hinB, const float* hinF,
                 float* houtF, bf16u* houtB){
    gemm_launch(stream, ACT_NONE, xb,   PP_, wih[g], PP_, bih[g], gi, H3_, nullptr, B_, H3_, H3_, PP_);
    gemm_launch(stream, ACT_NONE, hinB, H_,  whh[g], H_,  bhh[g], gh, H3_, nullptr, B_, H3_, H3_, H_);
    k_gru_combine<<<CEILDIV(B_*H_,EW),EW,0,stream>>>(gi, gh, hinF, houtF, houtB);
  };
  // gated message aggregation -> gmH
  auto gm = [&](const float* hidBase, int vc, int k, int nodeBase, int thr){
    size_t n = (size_t)B_*k*H_;
    k_build_hcat<<<(unsigned)CEILDIV(n,EW),EW,0,stream>>>(hidBase, adj, hcat, vc, k, nodeBase, thr);
    gemm_launch(stream, ACT_SIG,  hcat, H2_, wGate, H2_, bGate,   gpre, H_, nullptr, B_*k, H_, H_, H2_);
    gemm_launch(stream, ACT_NONE, hcat, H2_, wMap,  H2_, nullptr, mpre, H_, nullptr, B_*k, H_, H_, H2_);
    k_gm_reduce<<<CEILDIV(B_*H_,EW),EW,0,stream>>>(gpre, mpre, k, gmH, gmHB);
  };

  // ================= encoder =================
  for (int v = NN_-1; v >= 0; v--){
    const bf16u* hinB; const float* hinF;
    if (v == NN_-1) { hinB = hzB; hinF = hz; }
    else            { gm(henc, v, NN_-1-v, v+1, -1); hinB = gmHB; hinF = gmH; }
    buildX(v, 0);
    gru(xbuf, 0, hinB, hinF, tmpH, tmpHB);                      // pe
    buildX(v, 1);                                               // Xloop = X * adj[v][v]
    gru(xbuf, 1, tmpHB, tmpH, henc + (size_t)v*B_*H_, hencB + (size_t)v*B_*H_); // le
  }

  // ================= latent =================
  const bf16u* HgB = hencB;  // hid[0]
  gemm_launch(stream, ACT_NONE, HgB, H_, wMu,  H_, bMu,  muF,  L_, muB,     B_, L_, L_, H_);
  gemm_launch(stream, ACT_SP,   HgB, H_, wStd, H_, bStd, stdF, L_, nullptr, B_, L_, L_, H_);
  gemm_launch(stream, ACT_TANH, muB, L_, wZh,  L_, bZh,  hinit,H_, hinitB,  B_, H_, H_, L_);
  k_loss_kld<<<1,256,0,stream>>>(muF, stdF, acc);

  // ================= decoder init =================
  gru(xzero, 2, hinitB, hinit, tmpH, tmpHB);                    // pd
  gru(xzero, 3, tmpHB,  tmpH,  hdec, hdecB);                    // ld -> hid2[0]

  auto prop = [&](int vi, int s, int smmode){
    gm(hdec, vi, vi, 0, vi - s);                                // masks set for kk >= vi-s
    buildX(vi, 0);
    gru(xbuf, 2, gmHB, gmH, tmpH, tmpHB);                       // pd
    float* hv  = hdec  + (size_t)vi*B_*H_;
    bf16u* hvB = hdecB + (size_t)vi*B_*H_;
    if (smmode == 0) { gru(xzero, 3, tmpHB, tmpH, hv, hvB); }   // ld, sm = 0
    else { buildX(vi, 1); gru(xbuf, 3, tmpHB, tmpH, hv, hvB); } // ld, sm = adj[vi][vi]
  };

  // ================= decoder loop =================
  for (int vi = 1; vi < NN_; vi++){
    const bf16u* hprevB = hdecB + (size_t)(vi-1)*B_*H_;
    // Xi = mlp(hid2[vi-1], p1, p2)
    gemm_launch(stream, ACT_RELU, hprevB, H_,  wP1, H_,  bP1, t1, H2_, t1B,    B_, H2_, H2_, H_);
    gemm_launch(stream, ACT_NONE, t1B,    H2_, wP2, H2_, bP2, xi, 16,  nullptr,B_, 128, P_,  H2_);
    k_loss_lp<<<1,256,0,stream>>>(xi, params, vi, acc);

    prop(vi, 0, 0);                                             // zero masks, sm = 0
    bf16u* hvB = hdecB + (size_t)vi*B_*H_;

    // Es = sigmoid(mlp(Hi, es1, es2)) -- stash before Hi is recomputed
    gemm_launch(stream, ACT_RELU, hvB, H_,  wEs1, H_,  bEs1, t1,  H2_, t1B,    B_, H2_, H2_, H_);
    gemm_launch(stream, ACT_SIG,  t1B, H2_, wEs2, H2_, bEs2, esb, 16,  nullptr,B_, 128, 1,   H2_);

    prop(vi, 0, 1);                                             // zero masks, sm = self
    k_loss_es<<<1,256,0,stream>>>(esb, adj, vi, acc);

    for (int s = 1; s <= vi; s++){
      int vj = vi - s;
      // e = sigmoid(mlp(concat([Hi, hid2[vj]]), e1, e2))
      k_build_concat<<<CEILDIV(B_*H2_,EW),EW,0,stream>>>(hvB, hdecB + (size_t)vj*B_*H_, cc);
      gemm_launch(stream, ACT_RELU, cc,  H2_, wE1, H2_, bE1, t2, H4_, t2B,    B_, H4_, H4_, H2_);
      gemm_launch(stream, ACT_SIG,  t2B, H4_, wE2, H4_, bE2, eb, 16,  nullptr,B_, 128, 2,   H4_);
      k_loss_edge<<<1,256,0,stream>>>(eb, adj, vi, vj, acc);
      prop(vi, s, 1);                                           // masks include vj..vi-1
    }
  }

  k_finalize<<<1,32,0,stream>>>(acc, (float*)d_out);
}
